// NewsLevelMultiHeadSelfAttention_39410619908127
// MI455X (gfx1250) — matrix operand order repaired and checked by reading, not compiled, not run on hardware
//
#include <hip/hip_runtime.h>
#include <hip/hip_bf16.h>

typedef __attribute__((ext_vector_type(16))) _Float16 v16h;
typedef __attribute__((ext_vector_type(8)))  _Float16 v8h;
typedef __attribute__((ext_vector_type(4)))  _Float16 v4h;
typedef __attribute__((ext_vector_type(8)))  float    v8f;
typedef __attribute__((ext_vector_type(4)))  float    v4f;

#define B_  32
#define M_  512
#define D_  1024
#define H_  16
#define HD_ 64

static __device__ inline v16h cat8(v8h x, v8h y) {
  return __builtin_shufflevector(x, y, 0,1,2,3,4,5,6,7,8,9,10,11,12,13,14,15);
}

// load 8 contiguous f32 and round to 8 f16
static __device__ inline v8h cvt8(const float* __restrict__ p) {
  v4f x = *(const v4f*)p;
  v4f y = *(const v4f*)(p + 4);
  v8h r;
  #pragma unroll
  for (int j = 0; j < 4; ++j) { r[j] = (_Float16)x[j]; r[4+j] = (_Float16)y[j]; }
  return r;
}

// ---------------------------------------------------------------------------
// Stage A: Q/V projection GEMM.  Y = X @ W^T  (W is (out,in) row-major).
// grid = (BM/64 row tiles, D/128 col tiles, 2: 0=Q 1=V), block = 256 (8 waves).
// Wave w computes a 32x32 output tile (2x2 of 16x16) with A/B register reuse:
//   rows (w>>2)*32, cols (w&3)*32 within the 64x128 block.
// Q written head-major (B,H,M,64) f16; V written transposed (B,H,64,M) f16.
// ---------------------------------------------------------------------------
__global__ __launch_bounds__(256) void proj_kernel(
    const float* __restrict__ X,
    const float* __restrict__ WQ,
    const float* __restrict__ WV,
    _Float16* __restrict__ Qh,
    _Float16* __restrict__ Vt)
{
  const int tid  = threadIdx.x;
  const int w    = tid >> 5;
  const int lane = tid & 31;
  const int lo   = lane & 15;
  const int hi   = lane >> 4;

  const int isV = blockIdx.z;
  const float* Wm = isV ? WV : WQ;

  const int R0 = blockIdx.x * 64 + (w >> 2) * 32;       // 32-row wave tile base
  const int e0 = blockIdx.y * 128 + (w & 3) * 32;       // 32-col wave tile base

  const float* xrow0 = X  + (size_t)(R0 + lo)      * D_;
  const float* xrow1 = X  + (size_t)(R0 + 16 + lo) * D_;
  const float* wrow0 = Wm + (size_t)(e0 + lo)      * D_;
  const float* wrow1 = Wm + (size_t)(e0 + 16 + lo) * D_;

  v8f acc00 = {}, acc01 = {}, acc10 = {}, acc11 = {};

  for (int k0 = 0; k0 < D_; k0 += 32) {
    // A tiles (16x32 f16): halfs 0-7 = K[hi*8..], 8-15 = K[16+hi*8..]
    v16h A0 = cat8(cvt8(xrow0 + k0 + hi*8), cvt8(xrow0 + k0 + 16 + hi*8));
    v16h A1 = cat8(cvt8(xrow1 + k0 + hi*8), cvt8(xrow1 + k0 + 16 + hi*8));
    // B tiles (32x16 f16): lane = column, halfs 0-7 = K[hi*8..], 8-15 = K[16+hi*8..]
    v16h B0 = cat8(cvt8(wrow0 + k0 + hi*8), cvt8(wrow0 + k0 + 16 + hi*8));
    v16h B1 = cat8(cvt8(wrow1 + k0 + hi*8), cvt8(wrow1 + k0 + 16 + hi*8));

    acc00 = __builtin_amdgcn_wmma_f32_16x16x32_f16(false, A0, false, B0, (short)0, acc00, false, false);
    acc01 = __builtin_amdgcn_wmma_f32_16x16x32_f16(false, A0, false, B1, (short)0, acc01, false, false);
    acc10 = __builtin_amdgcn_wmma_f32_16x16x32_f16(false, A1, false, B0, (short)0, acc10, false, false);
    acc11 = __builtin_amdgcn_wmma_f32_16x16x32_f16(false, A1, false, B1, (short)0, acc11, false, false);
  }

  // D layout: lane holds column e, VGPR r holds strip-row (8*hi + r).
  const v8f* accs[4] = { &acc00, &acc01, &acc10, &acc11 };
  #pragma unroll
  for (int t = 0; t < 4; ++t) {
    const int Rstrip = R0 + (t >> 1) * 16;      // 16-row strip base (global)
    const int e      = e0 + (t & 1) * 16 + lo;  // output column for this lane
    const int b      = Rstrip >> 9;             // 512 rows per batch; no crossing
    const int mbase  = (Rstrip & 511) + 8 * hi;
    const int h      = e >> 6;
    const int dh     = e & 63;
    const v8f& a     = *accs[t];

    if (isV) {
      // Vt[b][h][dh][m] : m contiguous over r -> one 16B store of 8 halfs
      v8h pv;
      #pragma unroll
      for (int r = 0; r < 8; ++r) pv[r] = (_Float16)a[r];
      *(v8h*)(Vt + (((size_t)(b*H_ + h) * HD_ + dh) * M_ + mbase)) = pv;
    } else {
      // Qh[b][h][m][dh] : strided over r
      #pragma unroll
      for (int r = 0; r < 8; ++r)
        Qh[((size_t)(b*H_ + h) * M_ + (mbase + r)) * HD_ + dh] = (_Float16)a[r];
    }
  }
}

// ---------------------------------------------------------------------------
// K relayout: K is the UNPROJECTED input; cast f32 -> f16 into (B,H,M,64).
// ---------------------------------------------------------------------------
__global__ __launch_bounds__(256) void khcast_kernel(
    const float* __restrict__ X, _Float16* __restrict__ Kh)
{
  const unsigned q = (blockIdx.x * 256u + threadIdx.x) * 4u; // Kh flat index
  const int dh = q & 63;
  const int m  = (q >> 6) & 511;
  const int h  = (q >> 15) & 15;
  const int b  = q >> 19;
  v4f x = *(const v4f*)(X + ((size_t)(b*M_ + m)) * D_ + h*64 + dh);
  v4h o;
  #pragma unroll
  for (int j = 0; j < 4; ++j) o[j] = (_Float16)x[j];
  *(v4h*)(Kh + (size_t)q) = o;
}

// ---------------------------------------------------------------------------
// Stage B: flash attention per (b,h).  grid = (M/128, H, B), block = 256.
// Wave w owns 16 query rows; streams 32 key columns per iteration.
// S = Q K^T (4 wmma), online softmax (shfl_xor row reductions),
// P staged via per-wave LDS slice (D-layout -> A-layout), O += P V (4 wmma).
// ---------------------------------------------------------------------------
__global__ __launch_bounds__(256) void attn_kernel(
    const _Float16* __restrict__ Qh,
    const _Float16* __restrict__ Kh,
    const _Float16* __restrict__ Vt,
    float* __restrict__ Out)
{
  __shared__ _Float16 Plds[8][16][32];   // per-wave [row][col] P staging, 8 KB

  const int tid  = threadIdx.x;
  const int w    = tid >> 5;
  const int lane = tid & 31;
  const int lo   = lane & 15;
  const int hi   = lane >> 4;

  const int b  = blockIdx.z;
  const int h  = blockIdx.y;
  const int m0 = blockIdx.x * 128 + w * 16;

  const _Float16* Qp = Qh + (size_t)(b*H_ + h) * M_ * HD_;
  const _Float16* Kp = Kh + (size_t)(b*H_ + h) * M_ * HD_;
  const _Float16* Vp = Vt + (size_t)(b*H_ + h) * HD_ * M_;

  // Q strip in A layout, two k-chunks covering hd=64
  v16h AQ[2];
  {
    const _Float16* qrow = Qp + (size_t)(m0 + lo) * HD_;
    #pragma unroll
    for (int c = 0; c < 2; ++c) {
      v8h x = *(const v8h*)(qrow + c*32 + hi*8);
      v8h y = *(const v8h*)(qrow + c*32 + 16 + hi*8);
      AQ[c] = cat8(x, y);
    }
  }

  v8f acc[4] = {};
  float mrow[8], lrow[8];
  #pragma unroll
  for (int r = 0; r < 8; ++r) { mrow[r] = -INFINITY; lrow[r] = 0.0f; }

  const float L2E = 1.44269504088896340736f;

  for (int n0 = 0; n0 < M_; n0 += 32) {
    // prefetch next iteration's K rows and V columns into cache
    if (n0 + 32 < M_) {
      __builtin_prefetch(Kp + (size_t)(n0 + 32 + lane) * HD_, 0, 1);
      __builtin_prefetch(Vp + (size_t)(lane << 1) * M_ + n0 + 32, 0, 1);
    }

    // ---- S tiles: columns [n0,n0+16) and [n0+16,n0+32), K-dim hd=64
    v8f S0 = {}, S1 = {};
    #pragma unroll
    for (int c = 0; c < 2; ++c) {
      const _Float16* k0p = Kp + (size_t)(n0 + lo)      * HD_ + c*32 + hi*8;
      const _Float16* k1p = Kp + (size_t)(n0 + 16 + lo) * HD_ + c*32 + hi*8;
      v16h B0 = cat8(*(const v8h*)k0p, *(const v8h*)(k0p + 16));
      v16h B1 = cat8(*(const v8h*)k1p, *(const v8h*)(k1p + 16));
      S0 = __builtin_amdgcn_wmma_f32_16x16x32_f16(
          false, AQ[c], false, B0, (short)0, S0, false, false);
      S1 = __builtin_amdgcn_wmma_f32_16x16x32_f16(
          false, AQ[c], false, B1, (short)0, S1, false, false);
    }

    // ---- online softmax; row s = 8*hi + r lives across the 16-lane half
    #pragma unroll
    for (int r = 0; r < 8; ++r) {
      float s0 = S0[r], s1 = S1[r];
      float tm = fmaxf(s0, s1);
      tm = fmaxf(tm, __shfl_xor(tm, 1));
      tm = fmaxf(tm, __shfl_xor(tm, 2));
      tm = fmaxf(tm, __shfl_xor(tm, 4));
      tm = fmaxf(tm, __shfl_xor(tm, 8));
      const float mnew  = fmaxf(mrow[r], tm);
      const float scale = exp2f((mrow[r] - mnew) * L2E);
      const float p0    = exp2f((s0 - mnew) * L2E);
      const float p1    = exp2f((s1 - mnew) * L2E);
      float ts = p0 + p1;
      ts += __shfl_xor(ts, 1);
      ts += __shfl_xor(ts, 2);
      ts += __shfl_xor(ts, 4);
      ts += __shfl_xor(ts, 8);
      lrow[r] = lrow[r] * scale + ts;
      mrow[r] = mnew;
      acc[0][r] *= scale; acc[1][r] *= scale;
      acc[2][r] *= scale; acc[3][r] *= scale;
      Plds[w][8*hi + r][lo]      = (_Float16)p0;
      Plds[w][8*hi + r][16 + lo] = (_Float16)p1;
    }

    // LDS is in-order per wave; explicit wait as cheap insurance before the
    // cross-lane readback.
    asm volatile("s_wait_dscnt 0" ::: "memory");

    // ---- P readback in A layout (16 rows x 32 cols, k = column index)
    v16h AP;
    {
      const _Float16* pr = &Plds[w][lo][0];
      v8h x = *(const v8h*)(pr + hi*8);
      v8h y = *(const v8h*)(pr + 16 + hi*8);
      AP = cat8(x, y);
    }

    // ---- O += P * V : B columns d are rows of Vt (contiguous in n)
    #pragma unroll
    for (int t = 0; t < 4; ++t) {
      const _Float16* vr = Vp + (size_t)(t*16 + lo) * M_ + n0 + hi*8;
      v16h BV = cat8(*(const v8h*)vr, *(const v8h*)(vr + 16));
      acc[t] = __builtin_amdgcn_wmma_f32_16x16x32_f16(
          false, AP, false, BV, (short)0, acc[t], false, false);
    }
  }

  // ---- epilogue: scale by 1/l, scatter to (B,M,D) f32
  #pragma unroll
  for (int r = 0; r < 8; ++r) {
    const float inv = 1.0f / lrow[r];
    const int m = m0 + 8*hi + r;
    float* orow = Out + ((size_t)(b*M_ + m)) * D_ + h * HD_;
    #pragma unroll
    for (int t = 0; t < 4; ++t)
      orow[t*16 + lo] = acc[t][r] * inv;
  }
}

// ---------------------------------------------------------------------------
extern "C" void kernel_launch(void* const* d_in, const int* in_sizes, int n_in,
                              void* d_out, int out_size, void* d_ws, size_t ws_size,
                              hipStream_t stream) {
  const float* X  = (const float*)d_in[0];
  const float* WQ = (const float*)d_in[1];
  const float* WV = (const float*)d_in[2];
  float* Out = (float*)d_out;

  // f16 intermediates: Qh (B,H,M,64) 32MB | Vt (B,H,64,M) 32MB | Kh 32MB
  char* ws = (char*)d_ws;
  _Float16* Qh = (_Float16*)(ws);
  _Float16* Vt = (_Float16*)(ws + (size_t)33554432);
  _Float16* Kh = (_Float16*)(ws + (size_t)67108864);

  proj_kernel  <<<dim3((B_*M_)/64, D_/128, 2), 256, 0, stream>>>(X, WQ, WV, Qh, Vt);
  khcast_kernel<<<(B_*H_*M_*HD_)/(256*4),      256, 0, stream>>>(X, Kh);
  attn_kernel  <<<dim3(M_/128, H_, B_),        256, 0, stream>>>(Qh, Kh, Vt, Out);
}
